// windowSS2D_15985868275960
// MI455X (gfx1250) — matrix operand order repaired and checked by reading, not compiled, not run on hardware
//
#include <hip/hip_runtime.h>
#include <hip/hip_bf16.h>

// ---------------------------------------------------------------------------
// Problem constants
// ---------------------------------------------------------------------------
#define HH 64
#define WW 64
#define NPIX 4096          // 64*64
#define DMODEL 64
#define DINNER 128
#define OUTCH 64
#define DSTATE 16
#define DTRANK 4
#define NPROJ 144          // 4 * (4 + 16 + 16)
#define NXZ 192            // DINNER + OUTCH
#define KADD 1152          // DINNER * 9

typedef __attribute__((ext_vector_type(16))) _Float16 v16h;
typedef __attribute__((ext_vector_type(8)))  float    v8f;

// ---------------------------------------------------------------------------
// WMMA fragment loaders (wave32 layouts per CDNA5 ISA 7.12.2)
// A: 16x32 f16, row-major source, lane = hi*16 + (m&15)
//    elem j: K = (j>>3)*16 + hi*8 + (j&7)  -> two contiguous 8-half runs
// B: 32x16 f16, Bt[n][k] source (n-major), elem j: K = (j>>3)*16 + hi*8 + (j&7)
// ---------------------------------------------------------------------------
__device__ __forceinline__ v16h load_A_frag(const _Float16* A, int lda,
                                            int m0, int kk, int lane) {
  const int m  = m0 + (lane & 15);
  const int hi = lane >> 4;
  const _Float16* p0 = A + (size_t)m * lda + kk + hi * 8;
  v16h a;
  ((uint4*)&a)[0] = *(const uint4*)(p0);
  ((uint4*)&a)[1] = *(const uint4*)(p0 + 16);
  return a;
}

__device__ __forceinline__ v16h load_B_frag(const _Float16* Bt, int ldb,
                                            int n0, int kk, int lane) {
  const int n  = n0 + (lane & 15);
  const int hi = lane >> 4;
  const _Float16* p0 = Bt + (size_t)n * ldb + kk + hi * 8;
  v16h b;
  ((uint4*)&b)[0] = *(const uint4*)(p0);
  ((uint4*)&b)[1] = *(const uint4*)(p0 + 16);
  return b;
}

// ---------------------------------------------------------------------------
// Kernel 0: weight prep -> f16 (all weights are already Bt[n][k] layout)
// ---------------------------------------------------------------------------
__global__ void prep_weights(const float* in_proj_w, const float* x_proj_w,
                             const float* addconv_w, const float* out_proj_w,
                             const float* skip_w,
                             _Float16* W1t, _Float16* W2t, _Float16* W3t,
                             _Float16* W4t) {
  const int i = blockIdx.x * blockDim.x + threadIdx.x;
  if (i < NXZ * DMODEL)   W1t[i] = (_Float16)in_proj_w[i];
  if (i < NPROJ * DINNER) W2t[i] = (_Float16)x_proj_w[i];
  if (i < OUTCH * KADD)   W3t[i] = (_Float16)addconv_w[i];   // identical flat layout
  if (i < OUTCH * 128) {
    const int n = i >> 7, k = i & 127;
    W4t[i] = (_Float16)(k < 64 ? out_proj_w[n * 64 + k]
                               : skip_w[n * 64 + (k - 64)]);
  }
}

// ---------------------------------------------------------------------------
// Kernel 1: input layernorm -> xn (f16 GEMM A), raw x -> A2[:,64:128] (f16)
// one pixel per 64-thread block
// ---------------------------------------------------------------------------
__global__ void ln_in_kernel(const float* __restrict__ x,
                             const float* __restrict__ g,
                             const float* __restrict__ b,
                             _Float16* __restrict__ xn16,
                             _Float16* __restrict__ A2) {
  const int pix = blockIdx.x;
  const int c = threadIdx.x;
  __shared__ float s[DMODEL];
  const float v = x[pix * DMODEL + c];
  s[c] = v;
  __syncthreads();
  float mu = 0.f;
  #pragma unroll
  for (int i = 0; i < DMODEL; ++i) mu += s[i];
  mu *= (1.f / DMODEL);
  float var = 0.f;
  #pragma unroll
  for (int i = 0; i < DMODEL; ++i) { const float d = s[i] - mu; var += d * d; }
  var *= (1.f / DMODEL);
  const float xn = (v - mu) * rsqrtf(var + 1e-5f) * g[c] + b[c];
  xn16[pix * DMODEL + c] = (_Float16)xn;
  A2[pix * 128 + 64 + c] = (_Float16)v;   // raw x feeds the fused skip GEMM
}

// ---------------------------------------------------------------------------
// Kernel 2: xz = xn @ in_proj_w.T   (M=4096, K=64, N=192), f32 out
// ---------------------------------------------------------------------------
__global__ void gemm_xz(const _Float16* __restrict__ A,
                        const _Float16* __restrict__ Bt,
                        float* __restrict__ C) {
  const int wave = (blockIdx.x * blockDim.x + threadIdx.x) >> 5;
  const int lane = threadIdx.x & 31;
  const int TN = NXZ / 16;
  if (wave >= (NPIX / 16) * TN) return;
  const int tm = wave / TN, tn = wave % TN;
  v8f acc = {};
  #pragma unroll
  for (int kk = 0; kk < DMODEL; kk += 32) {
    v16h a = load_A_frag(A, DMODEL, tm * 16, kk, lane);
    v16h b = load_B_frag(Bt, DMODEL, tn * 16, kk, lane);
    acc = __builtin_amdgcn_wmma_f32_16x16x32_f16(false, a, false, b,
                                                 (short)0, acc, false, false);
  }
  const int col = tn * 16 + (lane & 15);
  const int row0 = tm * 16 + (lane >> 4) * 8;
  #pragma unroll
  for (int r = 0; r < 8; ++r) C[(size_t)(row0 + r) * NXZ + col] = acc[r];
}

// ---------------------------------------------------------------------------
// Kernel 3: depthwise 3x3 conv (zero pad) + bias + SiLU -> xc f32 + f16
// ---------------------------------------------------------------------------
__global__ void dwconv_kernel(const float* __restrict__ xz,
                              const float* __restrict__ cw,
                              const float* __restrict__ cb,
                              float* __restrict__ xc,
                              _Float16* __restrict__ xc16) {
  const int pix = blockIdx.x;
  const int d = threadIdx.x;               // 128 channels
  const int hh = pix >> 6, ww = pix & 63;
  float acc = cb[d];
  #pragma unroll
  for (int di = 0; di < 3; ++di) {
    const int h2 = hh + di - 1;
    if (h2 < 0 || h2 >= HH) continue;
    #pragma unroll
    for (int dj = 0; dj < 3; ++dj) {
      const int w2 = ww + dj - 1;
      if (w2 < 0 || w2 >= WW) continue;
      acc += xz[(size_t)(h2 * WW + w2) * NXZ + d] * cw[d * 9 + di * 3 + dj];
    }
  }
  const float s = acc / (1.f + __expf(-acc));   // silu
  xc[(size_t)pix * DINNER + d] = s;
  xc16[(size_t)pix * DINNER + d] = (_Float16)s;
}

// ---------------------------------------------------------------------------
// Kernel 4: proj = xc @ x_proj_w.T  (M=4096, K=128, N=144), f32 out
// ---------------------------------------------------------------------------
__global__ void gemm_proj(const _Float16* __restrict__ A,
                          const _Float16* __restrict__ Bt,
                          float* __restrict__ C) {
  const int wave = (blockIdx.x * blockDim.x + threadIdx.x) >> 5;
  const int lane = threadIdx.x & 31;
  const int TN = NPROJ / 16;
  if (wave >= (NPIX / 16) * TN) return;
  const int tm = wave / TN, tn = wave % TN;
  v8f acc = {};
  #pragma unroll
  for (int kk = 0; kk < DINNER; kk += 32) {
    v16h a = load_A_frag(A, DINNER, tm * 16, kk, lane);
    v16h b = load_B_frag(Bt, DINNER, tn * 16, kk, lane);
    acc = __builtin_amdgcn_wmma_f32_16x16x32_f16(false, a, false, b,
                                                 (short)0, acc, false, false);
  }
  const int col = tn * 16 + (lane & 15);
  const int row0 = tm * 16 + (lane >> 4) * 8;
  #pragma unroll
  for (int r = 0; r < 8; ++r) C[(size_t)(row0 + r) * NPROJ + col] = acc[r];
}

// ---------------------------------------------------------------------------
// Kernel 5: 4-direction, 9-step windowed selective scan + merge + layernorm
// one pixel per 128-thread block; thread = channel d; h[16] in registers.
// Output: A3[pix][d*9 + s] f16 (A-matrix for addconv GEMM, K ordered (d,i,j))
// ---------------------------------------------------------------------------
__global__ void scan_kernel(const float* __restrict__ xc,
                            const float* __restrict__ proj,
                            const float* __restrict__ dt_w,
                            const float* __restrict__ dt_b,
                            const float* __restrict__ A_logs,
                            const float* __restrict__ Ds,
                            const float* __restrict__ on_g,
                            const float* __restrict__ on_b,
                            _Float16* __restrict__ A3) {
  const int pix = blockIdx.x;
  const int d = threadIdx.x;               // 128
  const int hh = pix >> 6, ww = pix & 63;
  __shared__ float xcw[9][DINNER];         // xc window (reflect pad)
  __shared__ float pjw[9][NPROJ];          // proj window (reflect pad)
  __shared__ float ylds[9][DINNER];        // merged scan output
  __shared__ float mst[9], ist[9];

  int rp[9];
  #pragma unroll
  for (int p = 0; p < 9; ++p) {
    int h2 = hh + (p / 3) - 1; h2 = h2 < 0 ? -h2 : (h2 > 63 ? 126 - h2 : h2);
    int w2 = ww + (p % 3) - 1; w2 = w2 < 0 ? -w2 : (w2 > 63 ? 126 - w2 : w2);
    rp[p] = h2 * WW + w2;
  }
  #pragma unroll
  for (int p = 0; p < 9; ++p) xcw[p][d] = xc[(size_t)rp[p] * DINNER + d];
  for (int idx = d; idx < 9 * NPROJ; idx += 128) {
    const int p = idx / NPROJ, c = idx % NPROJ;
    pjw[p][c] = proj[(size_t)rp[p] * NPROJ + c];
  }
  float ycomb[9];
  #pragma unroll
  for (int s = 0; s < 9; ++s) ycomb[s] = 0.f;
  __syncthreads();

  for (int k = 0; k < 4; ++k) {
    // direction k uses proj channel group g: {0,2,1,3}
    const int g = ((k & 1) << 1) | (k >> 1);
    float Ad[DSTATE], dtw[DTRANK];
    #pragma unroll
    for (int n = 0; n < DSTATE; ++n)
      Ad[n] = -__expf(A_logs[(size_t)(k * DINNER + d) * DSTATE + n]);
    #pragma unroll
    for (int r = 0; r < DTRANK; ++r)
      dtw[r] = dt_w[(size_t)(k * DINNER + d) * DTRANK + r];
    const float dtb = dt_b[k * DINNER + d];
    const float Dd  = Ds[k * DINNER + d];
    float h[DSTATE];
    #pragma unroll
    for (int n = 0; n < DSTATE; ++n) h[n] = 0.f;

    for (int l = 0; l < 9; ++l) {
      const int q = (k & 2) ? (8 - l) : l;                 // dirs 2,3 reversed
      const int s = (k & 1) ? ((q % 3) * 3 + q / 3) : q;   // dirs 1,3 transposed
      const float* pr = &pjw[s][g * 36];
      float dts = dtb;
      #pragma unroll
      for (int r = 0; r < DTRANK; ++r) dts += pr[r] * dtw[r];
      const float delta = dts > 20.f ? dts : log1pf(__expf(dts));  // softplus
      const float u = xcw[s][d];
      const float du = delta * u;
      float y = 0.f;
      #pragma unroll
      for (int n = 0; n < DSTATE; ++n) {
        h[n] = h[n] * __expf(delta * Ad[n]) + du * pr[4 + n];  // B at c=4..19
        y += h[n] * pr[20 + n];                                // C at c=20..35
      }
      ycomb[s] += y + u * Dd;   // merge of 4 directions lands on spatial s
    }
  }

  #pragma unroll
  for (int s = 0; s < 9; ++s) ylds[s][d] = ycomb[s];
  __syncthreads();
  if (d < 9) {
    float mu = 0.f;
    for (int i = 0; i < DINNER; ++i) mu += ylds[d][i];
    mu *= (1.f / DINNER);
    float var = 0.f;
    for (int i = 0; i < DINNER; ++i) { const float t = ylds[d][i] - mu; var += t * t; }
    var *= (1.f / DINNER);
    mst[d] = mu;
    ist[d] = rsqrtf(var + 1e-5f);
  }
  __syncthreads();
  const float gg = on_g[d], bb = on_b[d];
  #pragma unroll
  for (int s = 0; s < 9; ++s) {
    const float v = (ylds[s][d] - mst[s]) * ist[s] * gg + bb;
    A3[(size_t)pix * KADD + d * 9 + s] = (_Float16)v;
  }
}

// ---------------------------------------------------------------------------
// Kernel 6: addconv GEMM (M=4096, K=1152, N=64) fused with bias + SiLU(z)
// gate; writes f16 into A2[:,0:64] (A-matrix of final GEMM)
// ---------------------------------------------------------------------------
__global__ void gemm_addconv(const _Float16* __restrict__ A,
                             const _Float16* __restrict__ Bt,
                             const float* __restrict__ xz,
                             const float* __restrict__ addconv_b,
                             _Float16* __restrict__ A2) {
  const int wave = (blockIdx.x * blockDim.x + threadIdx.x) >> 5;
  const int lane = threadIdx.x & 31;
  const int TN = OUTCH / 16;
  if (wave >= (NPIX / 16) * TN) return;
  const int tm = wave / TN, tn = wave % TN;
  v8f acc = {};
  for (int kk = 0; kk < KADD; kk += 32) {
    v16h a = load_A_frag(A, KADD, tm * 16, kk, lane);
    v16h b = load_B_frag(Bt, KADD, tn * 16, kk, lane);
    acc = __builtin_amdgcn_wmma_f32_16x16x32_f16(false, a, false, b,
                                                 (short)0, acc, false, false);
  }
  const int col = tn * 16 + (lane & 15);
  const int row0 = tm * 16 + (lane >> 4) * 8;
  const float bias = addconv_b[col];
  #pragma unroll
  for (int r = 0; r < 8; ++r) {
    const int row = row0 + r;
    const float z = xz[(size_t)row * NXZ + DINNER + col];
    const float gate = z / (1.f + __expf(-z));          // silu(z)
    A2[(size_t)row * 128 + col] = (_Float16)((acc[r] + bias) * gate);
  }
}

// ---------------------------------------------------------------------------
// Kernel 7: final GEMM: [gated | x] @ [out_proj_w ; skip_w].T + skip_b
// (M=4096, K=128, N=64), f32 out -> d_out
// ---------------------------------------------------------------------------
__global__ void gemm_final(const _Float16* __restrict__ A,
                           const _Float16* __restrict__ Bt,
                           const float* __restrict__ skip_b,
                           float* __restrict__ out) {
  const int wave = (blockIdx.x * blockDim.x + threadIdx.x) >> 5;
  const int lane = threadIdx.x & 31;
  const int TN = OUTCH / 16;
  if (wave >= (NPIX / 16) * TN) return;
  const int tm = wave / TN, tn = wave % TN;
  v8f acc = {};
  #pragma unroll
  for (int kk = 0; kk < 128; kk += 32) {
    v16h a = load_A_frag(A, 128, tm * 16, kk, lane);
    v16h b = load_B_frag(Bt, 128, tn * 16, kk, lane);
    acc = __builtin_amdgcn_wmma_f32_16x16x32_f16(false, a, false, b,
                                                 (short)0, acc, false, false);
  }
  const int col = tn * 16 + (lane & 15);
  const int row0 = tm * 16 + (lane >> 4) * 8;
  const float bias = skip_b[col];
  #pragma unroll
  for (int r = 0; r < 8; ++r)
    out[(size_t)(row0 + r) * OUTCH + col] = acc[r] + bias;
}

// ---------------------------------------------------------------------------
// Host launcher
// ---------------------------------------------------------------------------
extern "C" void kernel_launch(void* const* d_in, const int* in_sizes, int n_in,
                              void* d_out, int out_size, void* d_ws, size_t ws_size,
                              hipStream_t stream) {
  const float* x          = (const float*)d_in[0];
  const float* in_norm_g  = (const float*)d_in[1];
  const float* in_norm_b  = (const float*)d_in[2];
  const float* in_proj_w  = (const float*)d_in[3];
  const float* conv_w     = (const float*)d_in[4];
  const float* conv_b     = (const float*)d_in[5];
  const float* x_proj_w   = (const float*)d_in[6];
  const float* dt_w       = (const float*)d_in[7];
  const float* dt_b       = (const float*)d_in[8];
  const float* A_logs     = (const float*)d_in[9];
  const float* Ds         = (const float*)d_in[10];
  const float* out_norm_g = (const float*)d_in[11];
  const float* out_norm_b = (const float*)d_in[12];
  const float* addconv_w  = (const float*)d_in[13];
  const float* addconv_b  = (const float*)d_in[14];
  const float* skip_w     = (const float*)d_in[15];
  const float* skip_b     = (const float*)d_in[16];
  const float* out_proj_w = (const float*)d_in[17];
  float* out = (float*)d_out;

  // workspace layout (256B aligned slices)
  char* base = (char*)d_ws;
  size_t off = 0;
  auto take = [&](size_t bytes) -> char* {
    char* p = base + off;
    off += (bytes + 255) & ~(size_t)255;
    return p;
  };
  _Float16* W1t  = (_Float16*)take((size_t)NXZ * DMODEL * 2);
  _Float16* W2t  = (_Float16*)take((size_t)NPROJ * DINNER * 2);
  _Float16* W3t  = (_Float16*)take((size_t)OUTCH * KADD * 2);
  _Float16* W4t  = (_Float16*)take((size_t)OUTCH * 128 * 2);
  _Float16* xn16 = (_Float16*)take((size_t)NPIX * DMODEL * 2);
  _Float16* A2   = (_Float16*)take((size_t)NPIX * 128 * 2);
  float*    xz   = (float*)take((size_t)NPIX * NXZ * 4);
  float*    xc   = (float*)take((size_t)NPIX * DINNER * 4);
  _Float16* xc16 = (_Float16*)take((size_t)NPIX * DINNER * 2);
  float*    proj = (float*)take((size_t)NPIX * NPROJ * 4);
  _Float16* A3   = (_Float16*)take((size_t)NPIX * KADD * 2);
  (void)ws_size; (void)in_sizes; (void)n_in; (void)out_size;

  // 0: weight conversion
  prep_weights<<<(OUTCH * KADD + 255) / 256, 256, 0, stream>>>(
      in_proj_w, x_proj_w, addconv_w, out_proj_w, skip_w, W1t, W2t, W3t, W4t);
  // 1: input layernorm
  ln_in_kernel<<<NPIX, DMODEL, 0, stream>>>(x, in_norm_g, in_norm_b, xn16, A2);
  // 2: in_proj GEMM (3072 tiles, 8 waves/block)
  gemm_xz<<<(NPIX / 16) * (NXZ / 16) / 8, 256, 0, stream>>>(xn16, W1t, xz);
  // 3: depthwise conv + silu
  dwconv_kernel<<<NPIX, DINNER, 0, stream>>>(xz, conv_w, conv_b, xc, xc16);
  // 4: x_proj GEMM (2304 tiles)
  gemm_proj<<<(NPIX / 16) * (NPROJ / 16) / 8, 256, 0, stream>>>(xc16, W2t, proj);
  // 5: selective scan + merge + out layernorm
  scan_kernel<<<NPIX, DINNER, 0, stream>>>(xc, proj, dt_w, dt_b, A_logs, Ds,
                                           out_norm_g, out_norm_b, A3);
  // 6: addconv GEMM + silu gate (1024 tiles)
  gemm_addconv<<<(NPIX / 16) * (OUTCH / 16) / 8, 256, 0, stream>>>(
      A3, W3t, xz, addconv_b, A2);
  // 7: fused out_proj + skip GEMM (1024 tiles)
  gemm_final<<<(NPIX / 16) * (OUTCH / 16) / 8, 256, 0, stream>>>(
      A2, W4t, skip_b, out);
}
